// SimpleModel_31705448579733
// MI455X (gfx1250) — compile-verified, matrix operand order checked by reading, not run on hardware
//
#include <hip/hip_runtime.h>
#include <hip/hip_bf16.h>
#include <math.h>

typedef unsigned long long u64;

#define TOLD 60000
#define TNEW 60000
#define NVERT 30000
#define MAXSTEPS 300
#define TKSIZE (1u << 17)
#define TKMASK (TKSIZE - 1u)
#define FTSIZE (1u << 19)
#define FTMASK (FTSIZE - 1u)
#define CHUNK 480              // old rows staged in LDS per pass (480 | 60000)
#define CTILES (CHUNK / 16)    // 30 WMMA tiles per chunk

typedef __attribute__((ext_vector_type(16))) _Float16 v16h;
typedef __attribute__((ext_vector_type(8)))  float    v8f;

union F16x16 { v16h v; _Float16 h[16]; int4 q[2]; };
union F32x8  { v8f  v; float    f[8];  };

// ---------------------------------------------------------------- utilities
__device__ __forceinline__ unsigned mix64(u64 k) {
  k ^= k >> 33; k *= 0xff51afd7ed558ccdULL;
  k ^= k >> 33; k *= 0xc4ceb9fe1a85ec53ULL;
  k ^= k >> 33;
  return (unsigned)k;
}

__device__ __forceinline__ void sort2(int& a, int& b) { if (a > b) { int t = a; a = b; b = t; } }
__device__ __forceinline__ void sort4(int& a, int& b, int& c, int& d) {
  sort2(a, b); sort2(c, d); sort2(a, c); sort2(b, d); sort2(b, c);
}
__device__ __forceinline__ void sort3(int& a, int& b, int& c) {
  sort2(a, b); sort2(a, c); sort2(b, c);
}

__device__ __forceinline__ float min_edge(const float* vp, int a, int b, int c, int d) {
  int id[4] = { a, b, c, d };
  float px[4], py[4], pz[4];
#pragma unroll
  for (int i = 0; i < 4; i++) { px[i] = vp[3*id[i]]; py[i] = vp[3*id[i]+1]; pz[i] = vp[3*id[i]+2]; }
  const int ea[6] = { 0,0,0,1,1,2 }, eb[6] = { 1,2,3,2,3,3 };
  float m = 3.4e38f;
#pragma unroll
  for (int e = 0; e < 6; e++) {
    float dx = px[ea[e]] - px[eb[e]];
    float dy = py[ea[e]] - py[eb[e]];
    float dz = pz[ea[e]] - pz[eb[e]];
    m = fminf(m, sqrtf(dx*dx + dy*dy + dz*dz));
  }
  return m;
}

// ---------------------------------------------------------------- init scratch
__global__ void k_init(u64* tkk, int* tkv, u64* ftab, unsigned* prio, int* nbr) {
  for (unsigned i = blockIdx.x * blockDim.x + threadIdx.x; i < FTSIZE;
       i += gridDim.x * blockDim.x) {
    ftab[i] = 0ULL;
    if (i < TKSIZE) { tkk[i] = 0ULL; tkv[i] = -1; }
    if (i < NVERT)  prio[i] = 0u;
    if (i < TOLD * 4) nbr[i] = -1;
  }
}

// ---------------------------------------------------------------- old-tet pass
// oh8 row (8 halves, 16B): (-2cx, -2cy, -2cz, 1, |c|^2, 0, 0, 0)
__global__ void k_old(const int* __restrict__ oi, const float* __restrict__ vp,
                      u64* tkk, int* tkv, u64* ftab, int* nbr,
                      float* v0o, float* tinv, int* degen,
                      _Float16* oh8, float* oel, unsigned* prio) {
  int t = blockIdx.x * blockDim.x + threadIdx.x;
  if (t >= TOLD) return;
  int a = oi[4*t], b = oi[4*t+1], c = oi[4*t+2], d = oi[4*t+3];

  // ---- exact-match key insert (sorted key)
  int s0 = a, s1 = b, s2 = c, s3 = d;
  sort4(s0, s1, s2, s3);
  u64 key = ((((u64)s0 * NVERT + (u64)s1) * NVERT + (u64)s2) * NVERT + (u64)s3);
  unsigned idx = mix64(key) & TKMASK;
  while (true) {
    u64 old = atomicCAS(&tkk[idx], 0ULL, key);
    if (old == 0ULL) { tkv[idx] = t; break; }
    if (old == key) break;               // duplicate key: first insert wins
    idx = (idx + 1u) & TKMASK;
  }

  // ---- geometry
  float x0 = vp[3*a], y0 = vp[3*a+1], z0 = vp[3*a+2];
  float e1x = vp[3*b]   - x0, e1y = vp[3*b+1] - y0, e1z = vp[3*b+2] - z0;
  float e2x = vp[3*c]   - x0, e2y = vp[3*c+1] - y0, e2z = vp[3*c+2] - z0;
  float e3x = vp[3*d]   - x0, e3y = vp[3*d+1] - y0, e3z = vp[3*d+2] - z0;
  float cxx = e2y*e3z - e2z*e3y;
  float cxy = e2z*e3x - e2x*e3z;
  float cxz = e2x*e3y - e2y*e3x;
  float det = e1x*cxx + e1y*cxy + e1z*cxz;
  int dg = (fabsf(det) < 1e-10f) ? 1 : 0;
  float m[9];
  if (dg) {
    m[0]=1.f;m[1]=0.f;m[2]=0.f; m[3]=0.f;m[4]=1.f;m[5]=0.f; m[6]=0.f;m[7]=0.f;m[8]=1.f;
  } else {
    float inv = 1.0f / det;
    m[0] = cxx*inv; m[1] = cxy*inv; m[2] = cxz*inv;                 // (e2 x e3)/det
    m[3] = (e3y*e1z - e3z*e1y)*inv; m[4] = (e3z*e1x - e3x*e1z)*inv; // (e3 x e1)/det
    m[5] = (e3x*e1y - e3y*e1x)*inv;
    m[6] = (e1y*e2z - e1z*e2y)*inv; m[7] = (e1z*e2x - e1x*e2z)*inv; // (e1 x e2)/det
    m[8] = (e1x*e2y - e1y*e2x)*inv;
  }
  v0o[3*t] = x0; v0o[3*t+1] = y0; v0o[3*t+2] = z0;
#pragma unroll
  for (int i = 0; i < 9; i++) tinv[9*t + i] = m[i];
  degen[t] = dg;

  // ---- fused-distance B row + min edge
  float cx = (x0 + vp[3*b] + vp[3*c] + vp[3*d]) * 0.25f;
  float cy = (y0 + vp[3*b+1] + vp[3*c+1] + vp[3*d+1]) * 0.25f;
  float cz = (z0 + vp[3*b+2] + vp[3*c+2] + vp[3*d+2]) * 0.25f;
  float c2 = cx*cx + cy*cy + cz*cz;
  oh8[8*t+0] = (_Float16)(-2.0f * cx);
  oh8[8*t+1] = (_Float16)(-2.0f * cy);
  oh8[8*t+2] = (_Float16)(-2.0f * cz);
  oh8[8*t+3] = (_Float16)1.0f;
  oh8[8*t+4] = (_Float16)c2;
  oh8[8*t+5] = (_Float16)0.0f;
  oh8[8*t+6] = (_Float16)0.0f;
  oh8[8*t+7] = (_Float16)0.0f;
  oel[t] = min_edge(vp, a, b, c, d);

  // ---- vertex -> tet priority
  int vv[4] = { a, b, c, d };
#pragma unroll
  for (int vi = 0; vi < 4; vi++)
    atomicMax(&prio[vv[vi]], (unsigned)(vi * TOLD + t) + 1u);

  // ---- face adjacency via packed-u64 hash (key<<19 | tet<<2 | local)
  const int FOP[4][3] = { {1,2,3}, {0,2,3}, {0,1,3}, {0,1,2} };
#pragma unroll
  for (int j = 0; j < 4; j++) {
    int f0 = vv[FOP[j][0]], f1 = vv[FOP[j][1]], f2 = vv[FOP[j][2]];
    sort3(f0, f1, f2);
    u64 fk = ((u64)f0 * NVERT + (u64)f1) * NVERT + (u64)f2;   // < 2^45
    u64 ent = (fk << 19) | ((u64)t << 2) | (u64)j;
    unsigned fi = mix64(fk) & FTMASK;
    while (true) {
      u64 old = atomicCAS(&ftab[fi], 0ULL, ent);
      if (old == 0ULL) break;
      if ((old >> 19) == fk) {
        int pt = (int)((old >> 2) & 0x1FFFFULL);
        int pl = (int)(old & 3ULL);
        nbr[pt*4 + pl] = t;
        nbr[t*4 + j]  = pt;
        break;
      }
      fi = (fi + 1u) & FTMASK;
    }
  }
}

// degenerate rows lose their neighbors (after all pairing completed)
__global__ void k_degen(const int* degen, int* nbr) {
  int t = blockIdx.x * blockDim.x + threadIdx.x;
  if (t >= TOLD) return;
  if (degen[t]) {
#pragma unroll
    for (int j = 0; j < 4; j++) nbr[4*t + j] = -1;
  }
}

// ---------------------------------------------------------------- new-tet pass
// nh8 row (8 halves, 16B): (cx, cy, cz, |c|^2, 1, 0, 0, 0); row TNEW = zero pad
__global__ void k_new(const int* __restrict__ ni, const float* __restrict__ vp,
                      const u64* tkk, const int* tkv, const unsigned* prio,
                      int* matched, float* ncent, _Float16* nh8,
                      float* nel, int* seed) {
  int t = blockIdx.x * blockDim.x + threadIdx.x;
  if (t >= TNEW) return;
  int a = ni[4*t], b = ni[4*t+1], c = ni[4*t+2], d = ni[4*t+3];
  int s0 = a, s1 = b, s2 = c, s3 = d;
  sort4(s0, s1, s2, s3);
  u64 key = ((((u64)s0 * NVERT + (u64)s1) * NVERT + (u64)s2) * NVERT + (u64)s3);
  int mt = -1;
  unsigned idx = mix64(key) & TKMASK;
  while (true) {
    u64 k = tkk[idx];
    if (k == 0ULL) break;
    if (k == key) { mt = tkv[idx]; break; }
    idx = (idx + 1u) & TKMASK;
  }
  matched[t] = mt;

  float cx = (vp[3*a] + vp[3*b] + vp[3*c] + vp[3*d]) * 0.25f;
  float cy = (vp[3*a+1] + vp[3*b+1] + vp[3*c+1] + vp[3*d+1]) * 0.25f;
  float cz = (vp[3*a+2] + vp[3*b+2] + vp[3*c+2] + vp[3*d+2]) * 0.25f;
  ncent[3*t] = cx; ncent[3*t+1] = cy; ncent[3*t+2] = cz;
  nh8[8*t+0] = (_Float16)cx;
  nh8[8*t+1] = (_Float16)cy;
  nh8[8*t+2] = (_Float16)cz;
  nh8[8*t+3] = (_Float16)(cx*cx + cy*cy + cz*cz);
  nh8[8*t+4] = (_Float16)1.0f;
  nh8[8*t+5] = (_Float16)0.0f;
  nh8[8*t+6] = (_Float16)0.0f;
  nh8[8*t+7] = (_Float16)0.0f;
  nel[t] = min_edge(vp, a, b, c, d);

  unsigned pr = prio[a];
  seed[t] = (pr > 0u) ? (int)((pr - 1u) % (unsigned)TOLD) : 0;

  if (t == 0) {                       // zero pad row used by hi/invalid lanes for A
#pragma unroll
    for (int i = 0; i < 8; i++) nh8[8*TNEW + i] = (_Float16)0.0f;
  }
}

// ---------------------------------------------------------------- barycentric walk
__global__ void k_walk(const int* matched, const float* ncent,
                       const float* v0o, const float* tinv, const int* nbr,
                       const int* seed, int* rem, int* act) {
  int t = blockIdx.x * blockDim.x + threadIdx.x;
  if (t >= TNEW) return;
  if (matched[t] >= 0) { rem[t] = 0; act[t] = 0; return; }
  float cx = ncent[3*t], cy = ncent[3*t+1], cz = ncent[3*t+2];
  int cur = seed[t];
  int active = 1, rr = cur;
  for (int s = 0; s < MAXSTEPS; s++) {
    float rx = cx - v0o[3*cur], ry = cy - v0o[3*cur+1], rz = cz - v0o[3*cur+2];
    const float* M = &tinv[9*cur];
    float b1 = M[0]*rx + M[1]*ry + M[2]*rz;
    float b2 = M[3]*rx + M[4]*ry + M[5]*rz;
    float b3 = M[6]*rx + M[7]*ry + M[8]*rz;
    float b0 = 1.0f - (b1 + b2 + b3);
    float ab[4] = { b0, b1, b2, b3 };
    float mn = ab[0]; int j = 0;
#pragma unroll
    for (int i = 1; i < 4; i++) if (ab[i] < mn) { mn = ab[i]; j = i; }
    if (mn >= -1e-4f) { rr = cur; active = 0; break; }
    int nb = nbr[4*cur + j];
    if (nb < 0) { rr = cur; active = 0; break; }
    cur = nb;
  }
  if (active) rr = cur;          // overwritten by fallback result anyway
  rem[t] = rr; act[t] = active;
}

// ---------------------------------------------------------------- WMMA nearest-centroid fallback
// Distance fully fused into the WMMA (K=5 of 32):
//   A row M = (x, y, z, p2, 1),  B col N = (-2x', -2y', -2z', 1, c2)
//   => C[M][N] = p2[M] - 2*dot + c2[N] = ||cent_M - old_N||^2
// 8 waves/block; old rows staged through LDS in 480-row chunks (b128 in, ds b128 out).
// No early returns: tail waves use zeroed A pad row and skip the store (barrier-safe).
__global__ __launch_bounds__(256) void k_fb(const _Float16* __restrict__ nh8,
                                            const _Float16* __restrict__ oh8,
                                            int* __restrict__ fb) {
  __shared__ int4 sB4[CHUNK + 1];                 // +1 = 16B zero slot for hi lanes
  int tid  = threadIdx.x;
  int wv   = tid >> 5;
  int lane = tid & 31;
  bool hi  = lane >= 16;
  int  lg  = lane & 15;
  int  mbase = (blockIdx.x * 8 + wv) * 16;
  bool valid = (mbase < TNEW);

  int4 z4; z4.x = 0; z4.y = 0; z4.z = 0; z4.w = 0;
  if (tid == 0) sB4[CHUNK] = z4;                  // zero fragment slot

  // A fragment: lanes 0-15 supply K=0..7 of row M; hi/invalid lanes -> zero pad row
  F16x16 A;
  {
    int Mrow = (valid && !hi) ? (mbase + lg) : TNEW;
    A.q[0] = ((const int4*)(nh8 + 8 * Mrow))[0];
    A.q[1] = z4;
  }

  float bd[8]; int bi[8];
#pragma unroll
  for (int r = 0; r < 8; r++) { bd[r] = 3.4e38f; bi[r] = 0; }

  const int4* gB = (const int4*)oh8;
  for (int tb = 0; tb < TOLD; tb += CHUNK) {
    __syncthreads();                              // prev chunk consumed / zero slot ready
    for (int i = tid; i < CHUNK; i += 256)        // coalesced b128 cooperative stage
      sB4[i] = gB[tb + i];
    __builtin_prefetch((const void*)(oh8 + 8 * (tb + CHUNK)), 0, 1);  // next chunk -> L2
    __syncthreads();

    for (int n = 0; n < CTILES; n += 2) {
      F16x16 B0, B1;
      int o0 = hi ? CHUNK : (n * 16 + lg);
      int o1 = hi ? CHUNK : ((n + 1) * 16 + lg);
      B0.q[0] = sB4[o0]; B0.q[1] = z4;            // ds_load_b128
      B1.q[0] = sB4[o1]; B1.q[1] = z4;
      F32x8 C0, C1;
#pragma unroll
      for (int r = 0; r < 8; r++) { C0.f[r] = 0.0f; C1.f[r] = 0.0f; }
      C0.v = __builtin_amdgcn_wmma_f32_16x16x32_f16(false, A.v, false, B0.v,
                                                    (short)0, C0.v, false, false);
      C1.v = __builtin_amdgcn_wmma_f32_16x16x32_f16(false, A.v, false, B1.v,
                                                    (short)0, C1.v, false, false);
      int N0 = tb + n * 16 + lg;
      int N1 = N0 + 16;
#pragma unroll
      for (int r = 0; r < 8; r++) {
        if (C0.f[r] < bd[r]) { bd[r] = C0.f[r]; bi[r] = N0; }
        if (C1.f[r] < bd[r]) { bd[r] = C1.f[r]; bi[r] = N1; }
      }
    }
  }

  // argmin across the 16 lanes of each half (prefer lower index on ties)
#pragma unroll
  for (int msk = 1; msk < 16; msk <<= 1) {
#pragma unroll
    for (int r = 0; r < 8; r++) {
      float od = __shfl_xor(bd[r], msk, 32);
      int   oi = __shfl_xor(bi[r], msk, 32);
      if (od < bd[r] || (od == bd[r] && oi < bi[r])) { bd[r] = od; bi[r] = oi; }
    }
  }
  if (valid && lg == 0) {
#pragma unroll
    for (int r = 0; r < 8; r++) fb[mbase + (hi ? 8 : 0) + r] = bi[r];
  }
}

// ---------------------------------------------------------------- final combine
__global__ void k_final(const int* __restrict__ ni, const int* __restrict__ oi,
                        const float* __restrict__ occ, const float* __restrict__ ncc,
                        const int* matched, const int* nbr,
                        const int* rem, const int* act, const int* fb,
                        const float* oel, const float* nel,
                        float* __restrict__ out) {
  int t = blockIdx.x * blockDim.x + threadIdx.x;
  if (t >= TNEW) return;
  int m = matched[t];
  int rsel = act[t] ? fb[t] : rem[t];
  rsel = min(max(rsel, 0), TOLD - 1);
  int remap = (m >= 0) ? m : rsel;

  int cands[5];
  cands[0] = remap;
#pragma unroll
  for (int j = 0; j < 4; j++) {
    int nb = nbr[4*remap + j];
    cands[1 + j] = (nb >= 0) ? nb : remap;
  }
  if (m >= 0) {
#pragma unroll
    for (int k = 0; k < 5; k++) cands[k] = m;
  }

  int nv0 = ni[4*t], nv1 = ni[4*t+1], nv2 = ni[4*t+2], nv3 = ni[4*t+3];
  float nx = ncc[3*t], ny = ncc[3*t+1], nz = ncc[3*t+2];
  float nelv = fmaxf(nel[t], 1e-8f);

  float raw[5], s = 0.0f;
#pragma unroll
  for (int k = 0; k < 5; k++) {
    int c = cands[k];
    int ov = 0;
#pragma unroll
    for (int vi = 0; vi < 4; vi++) {
      int cv = oi[4*c + vi];
      ov += (cv == nv0 || cv == nv1 || cv == nv2 || cv == nv3) ? 1 : 0;
    }
    float dx = occ[3*c] - nx, dy = occ[3*c+1] - ny, dz = occ[3*c+2] - nz;
    float d2 = dx*dx + dy*dy + dz*dz;
    raw[k] = expf(2.0f * (float)ov) / (d2 + 1e-8f);
    s += raw[k];
  }
#pragma unroll
  for (int k = 0; k < 5; k++) {
    int c = cands[k];
    out[5*t + k] = (float)c;                                   // cands
    out[300000 + 5*t + k] = raw[k] / s;                        // weights
    float ds = oel[c] / nelv;
    out[600000 + 5*t + k] = fminf(fmaxf(ds, 0.1f), 10.0f);     // density_scale
  }
}

// ---------------------------------------------------------------- launch
extern "C" void kernel_launch(void* const* d_in, const int* in_sizes, int n_in,
                              void* d_out, int out_size, void* d_ws, size_t ws_size,
                              hipStream_t stream) {
  const int*   new_idx = (const int*)d_in[0];
  const int*   old_idx = (const int*)d_in[1];
  const float* old_cc  = (const float*)d_in[2];
  const float* new_cc  = (const float*)d_in[3];
  const float* vp      = (const float*)d_in[4];
  float* out = (float*)d_out;
  (void)in_sizes; (void)n_in; (void)out_size; (void)ws_size;

  char* base = (char*)d_ws;
  size_t off = 0;
  auto carve = [&](size_t bytes) -> void* {
    void* r = base + off;
    off = (off + bytes + 255) & ~(size_t)255;
    return r;
  };
  u64*      tkk    = (u64*)carve((size_t)TKSIZE * 8);
  int*      tkv    = (int*)carve((size_t)TKSIZE * 4);
  u64*      ftab   = (u64*)carve((size_t)FTSIZE * 8);
  int*      nbr    = (int*)carve((size_t)TOLD * 4 * 4);
  int*      match  = (int*)carve((size_t)TNEW * 4);
  float*    v0o    = (float*)carve((size_t)TOLD * 3 * 4);
  float*    tinv   = (float*)carve((size_t)TOLD * 9 * 4);
  int*      degen  = (int*)carve((size_t)TOLD * 4);
  _Float16* oh8    = (_Float16*)carve((size_t)TOLD * 8 * 2);
  float*    oel    = (float*)carve((size_t)TOLD * 4);
  unsigned* prio   = (unsigned*)carve((size_t)NVERT * 4);
  float*    ncent  = (float*)carve((size_t)TNEW * 3 * 4);
  _Float16* nh8    = (_Float16*)carve((size_t)(TNEW + 1) * 8 * 2);
  float*    nel    = (float*)carve((size_t)TNEW * 4);
  int*      seed   = (int*)carve((size_t)TNEW * 4);
  int*      remv   = (int*)carve((size_t)TNEW * 4);
  int*      actv   = (int*)carve((size_t)TNEW * 4);
  int*      fbv    = (int*)carve((size_t)TNEW * 4);

  const int B = 256;
  k_init<<<(FTSIZE + B - 1) / B, B, 0, stream>>>(tkk, tkv, ftab, prio, nbr);
  k_old<<<(TOLD + B - 1) / B, B, 0, stream>>>(old_idx, vp, tkk, tkv, ftab, nbr,
                                              v0o, tinv, degen, oh8, oel, prio);
  k_degen<<<(TOLD + B - 1) / B, B, 0, stream>>>(degen, nbr);
  k_new<<<(TNEW + B - 1) / B, B, 0, stream>>>(new_idx, vp, tkk, tkv, prio,
                                              match, ncent, nh8, nel, seed);
  k_walk<<<(TNEW + B - 1) / B, B, 0, stream>>>(match, ncent, v0o, tinv, nbr,
                                               seed, remv, actv);
  int fbBlocks = (TNEW / 16 + 7) / 8;            // 8 waves (16 rows each) per block
  k_fb<<<fbBlocks, 256, 0, stream>>>(nh8, oh8, fbv);
  k_final<<<(TNEW + B - 1) / B, B, 0, stream>>>(new_idx, old_idx, old_cc, new_cc,
                                                match, nbr, remv, actv, fbv,
                                                oel, nel, out);
}
